// CEABlock_27247272526358
// MI455X (gfx1250) — matrix operand order repaired and checked by reading, not compiled, not run on hardware
//
#include <hip/hip_runtime.h>
#include <hip/hip_bf16.h>
#include <math.h>

// ---------------- problem constants ----------------
#define B_     32
#define L_     320
#define LT_    64
#define LS_    256
#define KEEP_  180
#define REM_   76
#define LNEW_  244      // LT_ + KEEP_
#define DIM_   768
#define HEADS_ 12
#define HD_    64
#define HID_   3072

typedef __bf16 bf16_t;
typedef __bf16 v16bf __attribute__((ext_vector_type(16)));
typedef __bf16 v8bf  __attribute__((ext_vector_type(8)));
typedef float  v8f   __attribute__((ext_vector_type(8)));

union Frag { v16bf v; v8bf h[2]; };

enum { MODE_BF16 = 0, MODE_RESF32 = 1, MODE_GELU = 2 };

// ---------------------------------------------------------------------------
// One wave computes a 32(M) x 64(N) tile, K unrolled by 64 with two fragment
// sets so loads for the next 32-K chunk are issued before the WMMAs consuming
// the current chunk (overlaps global fetch with matrix math).
//
// A-frag element e (lane l): K = k + (l>>4)*16 + e                [contiguous k order, same as B]
// B-frag element e (lane l): K = k + (l>>4)*16 + e                [contiguous k order, same as A]
// C tile (f32):  row = mBase + 16*msub + r + 8*(l>>4), col = nBase + 16*j + (l&15)
// Requires K % 64 == 0 (holds: 768, 3072, 320, 64).
// ---------------------------------------------------------------------------
struct FragSet { Frag a0, a1, b0, b1, b2, b3; };

__device__ __forceinline__ void load_set(FragSet& f,
                                         const bf16_t* __restrict__ aRow0,
                                         const bf16_t* __restrict__ aRow1,
                                         const bf16_t* __restrict__ bCol,
                                         long ldb, int lhi, int k) {
  f.a0.h[0] = *(const v8bf*)(aRow0 + k + lhi * 16);
  f.a0.h[1] = *(const v8bf*)(aRow0 + k + 8 + lhi * 16);
  f.a1.h[0] = *(const v8bf*)(aRow1 + k + lhi * 16);
  f.a1.h[1] = *(const v8bf*)(aRow1 + k + 8 + lhi * 16);
  f.b0.v = *(const v16bf*)(bCol + k);
  f.b1.v = *(const v16bf*)(bCol + (long)16 * ldb + k);
  f.b2.v = *(const v16bf*)(bCol + (long)32 * ldb + k);
  f.b3.v = *(const v16bf*)(bCol + (long)48 * ldb + k);
}

__device__ __forceinline__ void wmma_set(const FragSet& f, v8f (&acc)[2][4]) {
#define WM(m, n, bfr)                                                        \
  acc[m][n] = __builtin_amdgcn_wmma_f32_16x16x32_bf16(                       \
      false, (m ? f.a1.v : f.a0.v), false, f.bfr.v, (short)0, acc[m][n],     \
      false, false)
  WM(0, 0, b0); WM(0, 1, b1); WM(0, 2, b2); WM(0, 3, b3);
  WM(1, 0, b0); WM(1, 1, b1); WM(1, 2, b2); WM(1, 3, b3);
#undef WM
}

__device__ __forceinline__ void mm_k32x64(const bf16_t* __restrict__ aRow0,
                                          const bf16_t* __restrict__ aRow1,
                                          const bf16_t* __restrict__ bCol,
                                          long ldb, int lhi, int K,
                                          v8f (&acc)[2][4]) {
  FragSet s0, s1;
  load_set(s0, aRow0, aRow1, bCol, ldb, lhi, 0);
  for (int k = 0; k < K; k += 64) {
    load_set(s1, aRow0, aRow1, bCol, ldb, lhi, k + 32);  // prefetch 2nd half
    wmma_set(s0, acc);
    if (k + 64 < K) load_set(s0, aRow0, aRow1, bCol, ldb, lhi, k + 64);
    wmma_set(s1, acc);
  }
}

// ---------------- generic GEMM: C = A(MxK) * Bt^T(KxN) + bias, epilogues ----
__global__ __launch_bounds__(32) void gemm_wmma(
    const bf16_t* __restrict__ A, int lda,
    const bf16_t* __restrict__ Bt, int ldb,
    const float* __restrict__ bias,
    const float* __restrict__ resid,
    float* __restrict__ outF, bf16_t* __restrict__ outH, int ldo,
    int K, int mode) {
  const int lane = threadIdx.x & 31;
  const int lrow = lane & 15, lhi = lane >> 4;
  const int mBase = blockIdx.y << 5;
  const int nBase = blockIdx.x << 6;
  const bf16_t* aRow0 = A + (long)(mBase + lrow) * lda;
  const bf16_t* aRow1 = A + (long)(mBase + 16 + lrow) * lda;
  const bf16_t* bCol  = Bt + (long)(nBase + lrow) * ldb + lhi * 16;
  v8f acc[2][4] = {};
  mm_k32x64(aRow0, aRow1, bCol, ldb, lhi, K, acc);
#pragma unroll
  for (int m = 0; m < 2; ++m) {
#pragma unroll
    for (int j = 0; j < 4; ++j) {
#pragma unroll
      for (int r = 0; r < 8; ++r) {
        const int row = mBase + (m << 4) + r + (lhi << 3);
        const int col = nBase + (j << 4) + lrow;
        float v = acc[m][j][r] + bias[col];
        const long o = (long)row * ldo + col;
        if (mode == MODE_BF16) {
          outH[o] = (bf16_t)v;
        } else if (mode == MODE_RESF32) {
          outF[o] = v + resid[o];
        } else {  // exact GELU
          float g = 0.5f * v * (1.0f + erff(v * 0.70710678118654752f));
          outH[o] = (bf16_t)g;
        }
      }
    }
  }
}

// ---------------- attention: S = Q K^T * (1/8), per (b,h) -------------------
__global__ __launch_bounds__(32) void attn_qk(const bf16_t* __restrict__ Qh,
                                              const bf16_t* __restrict__ Kh,
                                              float* __restrict__ S) {
  const int lane = threadIdx.x & 31;
  const int lrow = lane & 15, lhi = lane >> 4;
  const long bh = blockIdx.z;
  const int mBase = blockIdx.y << 5;
  const int nBase = blockIdx.x << 6;
  const bf16_t* base = Qh + bh * (long)L_ * HD_;
  const bf16_t* aRow0 = base + (long)(mBase + lrow) * HD_;
  const bf16_t* aRow1 = base + (long)(mBase + 16 + lrow) * HD_;
  const bf16_t* bCol  = Kh + bh * (long)L_ * HD_ + (long)(nBase + lrow) * HD_ + lhi * 16;
  v8f acc[2][4] = {};
  mm_k32x64(aRow0, aRow1, bCol, HD_, lhi, HD_, acc);
  float* outp = S + bh * (long)L_ * L_;
#pragma unroll
  for (int m = 0; m < 2; ++m)
#pragma unroll
    for (int j = 0; j < 4; ++j)
#pragma unroll
      for (int r = 0; r < 8; ++r) {
        const int row = mBase + (m << 4) + r + (lhi << 3);
        const int col = nBase + (j << 4) + lrow;
        outp[(long)row * L_ + col] = acc[m][j][r] * 0.125f;  // HD^-0.5
      }
}

// ---------------- attention: O = P V, V given transposed (HD_ x L_) ---------
__global__ __launch_bounds__(32) void attn_av(const bf16_t* __restrict__ P,
                                              const bf16_t* __restrict__ Vt,
                                              bf16_t* __restrict__ ctx) {
  const int lane = threadIdx.x & 31;
  const int lrow = lane & 15, lhi = lane >> 4;
  const long bh = blockIdx.z;
  const int b = (int)(bh / HEADS_), h = (int)(bh % HEADS_);
  const int mBase = blockIdx.y << 5;
  const bf16_t* base = P + bh * (long)L_ * L_;
  const bf16_t* aRow0 = base + (long)(mBase + lrow) * L_;
  const bf16_t* aRow1 = base + (long)(mBase + 16 + lrow) * L_;
  const bf16_t* bCol  = Vt + bh * (long)HD_ * L_ + (long)lrow * L_ + lhi * 16;
  v8f acc[2][4] = {};
  mm_k32x64(aRow0, aRow1, bCol, L_, lhi, L_, acc);
#pragma unroll
  for (int m = 0; m < 2; ++m)
#pragma unroll
    for (int j = 0; j < 4; ++j)
#pragma unroll
      for (int r = 0; r < 8; ++r) {
        const int row = mBase + (m << 4) + r + (lhi << 3);
        const int col = (j << 4) + lrow;  // 0..63 within the head
        ctx[((long)b * L_ + row) * DIM_ + h * HD_ + col] = (bf16_t)acc[m][j][r];
      }
}

// ---------------- LayerNorm + cast to bf16 (row = one token) ----------------
__global__ __launch_bounds__(256) void ln_cast(const float* __restrict__ X,
                                               const float* __restrict__ w,
                                               const float* __restrict__ bb,
                                               bf16_t* __restrict__ out) {
  const long row = blockIdx.x;
  const float* x = X + row * (long)DIM_;
  const int tid = threadIdx.x;
  float v0 = x[tid], v1 = x[tid + 256], v2 = x[tid + 512];
  float s = v0 + v1 + v2;
  float q = v0 * v0 + v1 * v1 + v2 * v2;
  for (int o = 16; o > 0; o >>= 1) {
    s += __shfl_xor(s, o, 32);
    q += __shfl_xor(q, o, 32);
  }
  __shared__ float shs[8], shq[8];
  if ((tid & 31) == 0) { shs[tid >> 5] = s; shq[tid >> 5] = q; }
  __syncthreads();
  float S = 0.f, Q = 0.f;
  for (int i = 0; i < 8; ++i) { S += shs[i]; Q += shq[i]; }
  const float mu = S * (1.0f / DIM_);
  const float var = Q * (1.0f / DIM_) - mu * mu;
  const float rs = rsqrtf(var + 1e-5f);
  bf16_t* o = out + row * (long)DIM_;
  o[tid]       = (bf16_t)((v0 - mu) * rs * w[tid]       + bb[tid]);
  o[tid + 256] = (bf16_t)((v1 - mu) * rs * w[tid + 256] + bb[tid + 256]);
  o[tid + 512] = (bf16_t)((v2 - mu) * rs * w[tid + 512] + bb[tid + 512]);
}

// ---------------- softmax over a 320-wide score row (in-place f32 + bf16) ---
__global__ __launch_bounds__(256) void softmax_row(float* __restrict__ S,
                                                   bf16_t* __restrict__ P) {
  const long row = blockIdx.x;
  float* p = S + row * (long)L_;
  const int tid = threadIdx.x;
  __shared__ float sh[8];
  float v0 = p[tid];
  float v1 = (tid < 64) ? p[256 + tid] : -1e30f;
  float m = fmaxf(v0, v1);
  for (int o = 16; o > 0; o >>= 1) m = fmaxf(m, __shfl_xor(m, o, 32));
  if ((tid & 31) == 0) sh[tid >> 5] = m;
  __syncthreads();
  float mm = sh[0];
  for (int i = 1; i < 8; ++i) mm = fmaxf(mm, sh[i]);
  __syncthreads();
  float e0 = __expf(v0 - mm);
  float e1 = (tid < 64) ? __expf(v1 - mm) : 0.f;
  float s = e0 + e1;
  for (int o = 16; o > 0; o >>= 1) s += __shfl_xor(s, o, 32);
  if ((tid & 31) == 0) sh[tid >> 5] = s;
  __syncthreads();
  float ss = 0.f;
  for (int i = 0; i < 8; ++i) ss += sh[i];
  const float inv = 1.0f / ss;
  bf16_t* pp = P + row * (long)L_;
  p[tid] = e0 * inv;
  pp[tid] = (bf16_t)(e0 * inv);
  if (tid < 64) {
    p[256 + tid] = e1 * inv;
    pp[256 + tid] = (bf16_t)(e1 * inv);
  }
}

// ---------------- repack qkv rows -> Qh, Kh, Vt (head-major, V transposed) --
__global__ __launch_bounds__(256) void repack_qkv(const bf16_t* __restrict__ qkv,
                                                  bf16_t* __restrict__ Qh,
                                                  bf16_t* __restrict__ Kh,
                                                  bf16_t* __restrict__ Vt) {
  const long idx = (long)blockIdx.x * 256 + threadIdx.x;  // B_*L_*DIM_
  const int c = (int)(idx % DIM_);
  const long t = idx / DIM_;
  const int b = (int)(t / L_), tok = (int)(t % L_);
  const int h = c >> 6, d = c & 63;
  const bf16_t* src = qkv + t * (long)(3 * DIM_);
  const long bh = (long)(b * HEADS_ + h);
  Qh[(bh * L_ + tok) * HD_ + d] = src[c];
  Kh[(bh * L_ + tok) * HD_ + d] = src[DIM_ + c];
  Vt[(bh * HD_ + d) * L_ + tok] = src[2 * DIM_ + c];
}

// ---------------- weight cast + transpose: W(KxN) f32 -> Wt(NxK) bf16 -------
__global__ __launch_bounds__(256) void cast_transpose(const float* __restrict__ W,
                                                      bf16_t* __restrict__ Wt,
                                                      int K, int N) {
  const long idx = (long)blockIdx.x * 256 + threadIdx.x;
  if (idx >= (long)K * N) return;
  const int n = (int)(idx % N);
  const long k = idx / N;
  Wt[(long)n * K + k] = (bf16_t)W[idx];
}

// ---------------- attn_t[b][s] = mean over (h, q<64) of attn[b,h,q,64+s] ----
__global__ __launch_bounds__(256) void attn_mean(const float* __restrict__ attn,
                                                 float* __restrict__ at) {
  const int b = blockIdx.x, s = threadIdx.x;
  const float* base = attn + (long)b * HEADS_ * L_ * L_ + LT_ + s;
  float sum = 0.f;
  for (int h = 0; h < HEADS_; ++h)
    for (int q = 0; q < LT_; ++q)
      sum += base[((long)h * L_ + q) * L_];
  at[b * LS_ + s] = sum * (1.0f / (HEADS_ * LT_));
}

// ---------------- stable descending argsort + top-k split -------------------
__global__ __launch_bounds__(256) void rank_select(const float* __restrict__ at,
                                                   const int* __restrict__ gis,
                                                   int* __restrict__ keep,
                                                   int* __restrict__ removed,
                                                   int* __restrict__ orderWS) {
  const int b = blockIdx.x, i = threadIdx.x;
  __shared__ float v[LS_];
  __shared__ int ord[LS_];
  const float vi = at[b * LS_ + i];
  v[i] = vi;
  __syncthreads();
  int r = 0;
  for (int j = 0; j < LS_; ++j) {
    const float vj = v[j];
    r += (int)(vj > vi) | ((int)(vj == vi) & (int)(j < i));
  }
  ord[r] = i;
  __syncthreads();
  const int o = ord[i];
  orderWS[b * LS_ + i] = o;
  const int g = gis[b * LS_ + o];
  if (i < KEEP_) keep[b * KEEP_ + i] = g;
  else           removed[b * REM_ + (i - KEEP_)] = g;
}

// ---------------- gather pruned token rows ----------------------------------
__global__ __launch_bounds__(256) void gather_rows(const float* __restrict__ x1,
                                                   const int* __restrict__ order,
                                                   float* __restrict__ x2) {
  const long row = blockIdx.x;  // B_*LNEW_
  const int b = (int)(row / LNEW_), t = (int)(row % LNEW_);
  const long src = (t < LT_) ? ((long)b * L_ + t)
                             : ((long)b * L_ + LT_ + order[b * LS_ + (t - LT_)]);
  const float* in = x1 + src * DIM_;
  float* o = x2 + row * (long)DIM_;
  for (int i = threadIdx.x; i < DIM_; i += 256) o[i] = in[i];
}

__global__ __launch_bounds__(256) void copy_i32(const int* __restrict__ in,
                                                int* __restrict__ out, int n) {
  const int i = blockIdx.x * 256 + threadIdx.x;
  if (i < n) out[i] = in[i];
}

// ---------------------------------------------------------------------------
extern "C" void kernel_launch(void* const* d_in, const int* in_sizes, int n_in,
                              void* d_out, int out_size, void* d_ws, size_t ws_size,
                              hipStream_t stream) {
  (void)in_sizes; (void)n_in; (void)out_size; (void)ws_size;
  const float* X[2]   = {(const float*)d_in[0], (const float*)d_in[1]};
  const int*   GIT[2] = {(const int*)d_in[2], (const int*)d_in[3]};
  const int*   GIS[2] = {(const int*)d_in[4], (const int*)d_in[5]};
  const float* n1w = (const float*)d_in[6],  *n1b = (const float*)d_in[7];
  const float* qkvW = (const float*)d_in[8], *qkvB = (const float*)d_in[9];
  const float* projW = (const float*)d_in[10], *projB = (const float*)d_in[11];
  const float* n2w = (const float*)d_in[12], *n2b = (const float*)d_in[13];
  const float* fc1W = (const float*)d_in[14], *fc1B = (const float*)d_in[15];
  const float* fc2W = (const float*)d_in[16], *fc2B = (const float*)d_in[17];
  float* out = (float*)d_out;

  // output offsets (flat float/int32 elements, return-tuple order)
  const long offX[2]    = {0L,        45328384L};
  const long offGIT[2]  = {5996544L,  51324928L};
  const long offKEEP[2] = {5998592L,  51326976L};
  const long offREM[2]  = {6004352L,  51332736L};
  const long offATTN[2] = {6006784L,  51335168L};

  // carve workspace
  char* wp = (char*)d_ws;
  auto take = [&](size_t bytes) -> void* {
    void* r = (void*)wp;
    wp += (bytes + 255) & ~(size_t)255;
    return r;
  };
  bf16_t* wtQKV  = (bf16_t*)take((size_t)DIM_ * 3 * DIM_ * 2);
  bf16_t* wtPROJ = (bf16_t*)take((size_t)DIM_ * DIM_ * 2);
  bf16_t* wtFC1  = (bf16_t*)take((size_t)DIM_ * HID_ * 2);
  bf16_t* wtFC2  = (bf16_t*)take((size_t)HID_ * DIM_ * 2);
  bf16_t* lnbuf  = (bf16_t*)take((size_t)B_ * L_ * DIM_ * 2);
  bf16_t* qkvbuf = (bf16_t*)take((size_t)B_ * L_ * 3 * DIM_ * 2);
  bf16_t* Qh     = (bf16_t*)take((size_t)B_ * HEADS_ * L_ * HD_ * 2);
  bf16_t* Kh     = (bf16_t*)take((size_t)B_ * HEADS_ * L_ * HD_ * 2);
  bf16_t* Vt     = (bf16_t*)take((size_t)B_ * HEADS_ * HD_ * L_ * 2);
  bf16_t* Pbuf   = (bf16_t*)take((size_t)B_ * HEADS_ * L_ * L_ * 2);
  bf16_t* ctx    = (bf16_t*)take((size_t)B_ * L_ * DIM_ * 2);
  float*  x1     = (float*)take((size_t)B_ * L_ * DIM_ * 4);
  float*  x2     = (float*)take((size_t)B_ * LNEW_ * DIM_ * 4);
  bf16_t* mlpin  = (bf16_t*)take((size_t)B_ * LNEW_ * DIM_ * 2);
  bf16_t* hbuf   = (bf16_t*)take((size_t)B_ * LNEW_ * HID_ * 2);
  float*  attnT  = (float*)take((size_t)B_ * LS_ * 4);
  int*    orderW = (int*)take((size_t)B_ * LS_ * 4);

  // weight prep (shared by both streams)
  cast_transpose<<<(DIM_ * 3 * DIM_ + 255) / 256, 256, 0, stream>>>(qkvW, wtQKV, DIM_, 3 * DIM_);
  cast_transpose<<<(DIM_ * DIM_ + 255) / 256, 256, 0, stream>>>(projW, wtPROJ, DIM_, DIM_);
  cast_transpose<<<(DIM_ * HID_ + 255) / 256, 256, 0, stream>>>(fc1W, wtFC1, DIM_, HID_);
  cast_transpose<<<(HID_ * DIM_ + 255) / 256, 256, 0, stream>>>(fc2W, wtFC2, HID_, DIM_);

  for (int s = 0; s < 2; ++s) {
    float* attnF = out + offATTN[s];
    // 1) LN1 + bf16 cast
    ln_cast<<<B_ * L_, 256, 0, stream>>>(X[s], n1w, n1b, lnbuf);
    // 2) QKV GEMM (10240x768 @ 768x2304) -> bf16
    gemm_wmma<<<dim3((3 * DIM_) / 64, (B_ * L_) / 32), 32, 0, stream>>>(
        lnbuf, DIM_, wtQKV, DIM_, qkvB, nullptr, nullptr, qkvbuf, 3 * DIM_, DIM_, MODE_BF16);
    // 3) repack into Qh/Kh and transposed V
    repack_qkv<<<(B_ * L_ * DIM_) / 256, 256, 0, stream>>>(qkvbuf, Qh, Kh, Vt);
    // 4) scores -> d_out attn slot (scratch), then softmax in place + bf16 P
    attn_qk<<<dim3(L_ / 64, L_ / 32, B_ * HEADS_), 32, 0, stream>>>(Qh, Kh, attnF);
    softmax_row<<<B_ * HEADS_ * L_, 256, 0, stream>>>(attnF, Pbuf);
    // 5) context = P @ V  -> bf16 (B,L,DIM)
    attn_av<<<dim3(1, L_ / 32, B_ * HEADS_), 32, 0, stream>>>(Pbuf, Vt, ctx);
    // 6) proj GEMM + bias + residual(x) -> x1 f32
    gemm_wmma<<<dim3(DIM_ / 64, (B_ * L_) / 32), 32, 0, stream>>>(
        ctx, DIM_, wtPROJ, DIM_, projB, X[s], x1, nullptr, DIM_, DIM_, MODE_RESF32);
    // 7) candidate elimination
    attn_mean<<<B_, 256, 0, stream>>>(attnF, attnT);
    rank_select<<<B_, 256, 0, stream>>>(attnT, GIS[s], (int*)(out + offKEEP[s]),
                                        (int*)(out + offREM[s]), orderW);
    gather_rows<<<B_ * LNEW_, 256, 0, stream>>>(x1, orderW, x2);
    // 8) MLP: LN2 -> fc1+GELU -> fc2 + residual(x2) -> d_out x slot (f32)
    ln_cast<<<B_ * LNEW_, 256, 0, stream>>>(x2, n2w, n2b, mlpin);
    gemm_wmma<<<dim3(HID_ / 64, (B_ * LNEW_) / 32), 32, 0, stream>>>(
        mlpin, DIM_, wtFC1, DIM_, fc1B, nullptr, nullptr, hbuf, HID_, DIM_, MODE_GELU);
    gemm_wmma<<<dim3(DIM_ / 64, (B_ * LNEW_) / 32), 32, 0, stream>>>(
        hbuf, HID_, wtFC2, HID_, fc2B, x2, out + offX[s], nullptr, DIM_, HID_, MODE_RESF32);
    // 9) passthrough template indices
    copy_i32<<<(B_ * LT_ + 255) / 256, 256, 0, stream>>>(GIT[s], (int*)(out + offGIT[s]), B_ * LT_);
  }
}
